// LatticeLSTM_69509750718690
// MI455X (gfx1250) — matrix operand order repaired and checked by reading, not compiled, not run on hardware
//
#include <hip/hip_runtime.h>

// ---------- types ----------
typedef __attribute__((ext_vector_type(16))) __bf16 v16bf;
typedef __attribute__((ext_vector_type(8)))  float  v8f;
typedef __attribute__((ext_vector_type(4)))  unsigned int u32x4;

union Frag16 { v16bf v; u32x4 q[2]; };

__device__ __forceinline__ unsigned short f2bf(float f) {
  unsigned int u = __float_as_uint(f);
  u += 0x7fffu + ((u >> 16) & 1u);           // round-to-nearest-even
  return (unsigned short)(u >> 16);
}
__device__ __forceinline__ float sigmoid_(float v) { return 1.0f / (1.0f + __expf(-v)); }
__device__ __forceinline__ float tanh_fast(float x) {
  float e = __expf(2.0f * x);
  return (e - 1.0f) / (e + 1.0f);
}
__device__ __forceinline__ v8f vzero8() {
  v8f z;
  #pragma unroll
  for (int i = 0; i < 8; ++i) z[i] = 0.0f;
  return z;
}

// async global -> LDS copy, 16 bytes per lane (ASYNCcnt-tracked)
__device__ __forceinline__ void async_load_b128(unsigned lds_byte_off, const void* gaddr) {
  asm volatile("global_load_async_to_lds_b128 %0, %1, off"
               :: "v"(lds_byte_off), "v"((unsigned long long)gaddr)
               : "memory");
}

// ---------- prep: weight / bias packing ----------
// Wx^T bf16 [2048][320]: slots {ii, ig, io, lx}, K padded 300 -> 320 with zeros
__global__ void pack_wx(const float* __restrict__ w0, const float* __restrict__ w1,
                        const float* __restrict__ w2, const float* __restrict__ w3,
                        unsigned short* __restrict__ out) {
  int idx = blockIdx.x * 256 + threadIdx.x;
  if (idx >= 2048 * 320) return;
  int n = idx / 320, k = idx % 320;
  int slot = n >> 9, nn = n & 511;
  const float* w = (slot == 0) ? w0 : (slot == 1) ? w1 : (slot == 2) ? w2 : w3;
  float v = (k < 300) ? w[k * 512 + nn] : 0.0f;
  out[idx] = f2bf(v);
}

// Wh^T bf16 [2048][512]: slots {hi, hg, ho, zero}
__global__ void pack_wh(const float* __restrict__ w0, const float* __restrict__ w1,
                        const float* __restrict__ w2, unsigned short* __restrict__ out) {
  int idx = blockIdx.x * 256 + threadIdx.x;
  if (idx >= 2048 * 512) return;
  int n = idx / 512, k = idx % 512;
  int slot = n >> 9, nn = n & 511;
  float v = 0.0f;
  if (slot == 0) v = w0[k * 512 + nn];
  else if (slot == 1) v = w1[k * 512 + nn];
  else if (slot == 2) v = w2[k * 512 + nn];
  out[idx] = f2bf(v);
}

// w_lc^T bf16 [512][512]
__global__ void pack_wlc(const float* __restrict__ w, unsigned short* __restrict__ out) {
  int idx = blockIdx.x * 256 + threadIdx.x;
  if (idx >= 512 * 512) return;
  int n = idx / 512, k = idx % 512;
  out[idx] = f2bf(w[k * 512 + n]);
}

// fused bias [2048]: {b_ii+b_hi, b_ig+b_hg, b_io+b_ho, b_lx}
__global__ void pack_bias(const float* __restrict__ bii, const float* __restrict__ bhi,
                          const float* __restrict__ big, const float* __restrict__ bhg,
                          const float* __restrict__ bio, const float* __restrict__ bho,
                          const float* __restrict__ blx, float* __restrict__ out) {
  int col = blockIdx.x * 256 + threadIdx.x;
  if (col >= 2048) return;
  int slot = col >> 9, nn = col & 511;
  float v;
  if (slot == 0) v = bii[nn] + bhi[nn];
  else if (slot == 1) v = big[nn] + bhg[nn];
  else if (slot == 2) v = bio[nn] + bho[nn];
  else v = blx[nn];
  out[col] = v;
}

// ---------- prep: activation conversion ----------
// x [8192][300] fp32 -> xb [8192][320] bf16, zero padded
__global__ void conv_pad_x(const float* __restrict__ x, unsigned short* __restrict__ xb) {
  int idx = blockIdx.x * 256 + threadIdx.x;
  if (idx >= 8192 * 320) return;
  int row = idx / 320, k = idx % 320;
  float v = (k < 300) ? x[row * 300 + k] : 0.0f;
  xb[idx] = f2bf(v);
}

// generic fp32 -> bf16 (2 elements per thread, packed 32-bit stores)
__global__ void conv_bf16x2(const float* __restrict__ in, unsigned int* __restrict__ out,
                            long n2) {
  long i = (long)blockIdx.x * 256 + threadIdx.x;
  if (i >= n2) return;
  float a = in[2 * i], b = in[2 * i + 1];
  out[i] = (unsigned int)f2bf(a) | ((unsigned int)f2bf(b) << 16);
}

// ---------- 128x128 tile GEMM accumulate (bf16 WMMA, fp32 accum) ----------
// Ab: bf16 row-major [M][ldK]; Bt: bf16 packed [N][Ktot]. 256 threads = 8 waves
// arranged 4(row) x 2(col); each wave computes 32 rows x 64 cols = acc[2][4].
// Staging uses double-buffered LDS fed by global_load_async_to_lds_b128:
// next K-tile streams into the other buffer while WMMAs consume the current one.
__device__ __forceinline__ void gemm_accum_128x128(
    const unsigned short* __restrict__ Ab, long tileM, int ldK,
    const unsigned short* __restrict__ Bt, int tileN, int Ktot,
    unsigned short* ldsA, unsigned short* ldsB, v8f (&acc)[2][4]) {
  const int tid  = threadIdx.x;
  const int wave = tid >> 5;
  const int lane = tid & 31;
  const int r    = lane & 15;
  const int half = lane >> 4;
  const int wr   = wave >> 1;        // 0..3: 32-row group
  const int wc   = wave & 1;         // 0..1: 64-col group
  const int srow = tid >> 1;         // staging row 0..127
  const int skh  = (tid & 1) * 16;   // staging k-half

  const unsigned short* aptr = Ab + (tileM + srow) * (long)ldK + skh;
  const unsigned short* bptr = Bt + (long)(tileN + srow) * Ktot + skh;
  // LDS byte offsets of this thread's staging slice (flat addr low 32 bits = LDS offset)
  const unsigned aoff = (unsigned)(unsigned long long)(const void*)(ldsA + srow * 32 + skh);
  const unsigned boff = (unsigned)(unsigned long long)(const void*)(ldsB + srow * 32 + skh);

  // prologue: stream first K-tile into buffer 0
  async_load_b128(aoff,      aptr);
  async_load_b128(aoff + 16, aptr + 8);
  async_load_b128(boff,      bptr);
  async_load_b128(boff + 16, bptr + 8);

  for (int ko = 0; ko < Ktot; ko += 32) {
    const unsigned bufsel = ((ko >> 5) & 1) ? 8192u : 0u;   // current buffer (bytes)
    if (ko + 32 < Ktot) {
      const unsigned nsel = bufsel ^ 8192u;                 // prefetch target buffer
      aptr += 32; bptr += 32;
      async_load_b128(aoff + nsel,      aptr);
      async_load_b128(aoff + nsel + 16, aptr + 8);
      async_load_b128(boff + nsel,      bptr);
      async_load_b128(boff + nsel + 16, bptr + 8);
      // async loads complete in order: <=4 outstanding => current buffer landed
      asm volatile("s_wait_asynccnt 0x4" ::: "memory");
    } else {
      asm volatile("s_wait_asynccnt 0x0" ::: "memory");
    }
    __syncthreads();

    const unsigned short* A0 = ldsA + (bufsel >> 1);
    const unsigned short* B0 = ldsB + (bufsel >> 1);
    Frag16 a[2], b[4];
    #pragma unroll
    for (int i = 0; i < 2; ++i) {
      const unsigned short* ap = A0 + (wr * 32 + i * 16 + r) * 32;
      a[i].q[0] = *(const u32x4*)(ap + half * 8);
      a[i].q[1] = *(const u32x4*)(ap + 16 + half * 8);
    }
    #pragma unroll
    for (int s = 0; s < 4; ++s) {
      const unsigned short* bp = B0 + (wc * 64 + s * 16 + r) * 32;
      b[s].q[0] = *(const u32x4*)(bp + half * 8);
      b[s].q[1] = *(const u32x4*)(bp + 16 + half * 8);
    }
    #pragma unroll
    for (int i = 0; i < 2; ++i)
      #pragma unroll
      for (int s = 0; s < 4; ++s)
        acc[i][s] = __builtin_amdgcn_wmma_f32_16x16x32_bf16(
            false, a[i].v, false, b[s].v, (short)0, acc[i][s], false, false);
    __syncthreads();   // all waves done reading before this buffer is refilled
  }
}

// ---------- gate kernel: act[B,2048] = f(x@Wx + h0@Wh + bias) ----------
// slots: 0 -> i = sigmoid, 1 -> g = tanh, 2 -> o = sigmoid, 3 -> lx (raw)
__global__ void __launch_bounds__(256)
gate_kernel(const unsigned short* __restrict__ xb, const unsigned short* __restrict__ hb,
            const unsigned short* __restrict__ Wxt, const unsigned short* __restrict__ Wht,
            const float* __restrict__ bias_all, float* __restrict__ act) {
  __shared__ unsigned short ldsA[2 * 128 * 32];   // double-buffered 8KB tiles
  __shared__ unsigned short ldsB[2 * 128 * 32];
  const long tileM = (long)blockIdx.y * 128;
  const int  tileN = blockIdx.x * 128;

  v8f acc[2][4];
  #pragma unroll
  for (int i = 0; i < 2; ++i)
    #pragma unroll
    for (int s = 0; s < 4; ++s) acc[i][s] = vzero8();

  gemm_accum_128x128(xb, tileM, 320, Wxt, tileN, 320, ldsA, ldsB, acc);
  gemm_accum_128x128(hb, tileM, 512, Wht, tileN, 512, ldsA, ldsB, acc);

  const int wave = threadIdx.x >> 5, lane = threadIdx.x & 31;
  const int r = lane & 15, half = lane >> 4;
  const int wr = wave >> 1, wc = wave & 1;
  #pragma unroll
  for (int i = 0; i < 2; ++i) {
    #pragma unroll
    for (int s = 0; s < 4; ++s) {
      int col = tileN + wc * 64 + s * 16 + r;
      int slot = col >> 9;
      float bb = bias_all[col];
      #pragma unroll
      for (int j = 0; j < 8; ++j) {
        long row = tileM + wr * 32 + i * 16 + j + 8 * half;  // WMMA C: M = j + 8*half
        float v = acc[i][s][j] + bb;
        if (slot == 1)      v = tanh_fast(v);
        else if (slot != 3) v = sigmoid_(v);
        act[row * 2048 + col] = v;
      }
    }
  }
}

// ---------- edge kernel: GEMM [B*W,512]@[512,512] + fused ragged-softmax merge ----------
__global__ void __launch_bounds__(256)
edge_kernel(const unsigned short* __restrict__ cb, const float* __restrict__ cbew,
            const unsigned short* __restrict__ Wlct, const float* __restrict__ act,
            const float* __restrict__ b_lc, const int* __restrict__ num_word,
            float* __restrict__ out) {
  __shared__ unsigned short ldsA[2 * 128 * 32];
  __shared__ unsigned short ldsB[2 * 128 * 32];
  const long tileM = (long)blockIdx.y * 128;   // rows = b*8 + w
  const int  tileN = blockIdx.x * 128;

  v8f acc[2][4];
  #pragma unroll
  for (int i = 0; i < 2; ++i)
    #pragma unroll
    for (int s = 0; s < 4; ++s) acc[i][s] = vzero8();

  gemm_accum_128x128(cb, tileM, 512, Wlct, tileN, 512, ldsA, ldsB, acc);

  const int wave = threadIdx.x >> 5, lane = threadIdx.x & 31;
  const int r = lane & 15, half = lane >> 4;
  const int wr = wave >> 1, wc = wave & 1;

  #pragma unroll
  for (int i = 0; i < 2; ++i) {
    // 16-row subtile rows = batches (base>>3) and (base>>3)+1, each with w=0..7
    const long base = tileM + wr * 32 + i * 16;
    const long b = (base >> 3) + half;
    const int nw = num_word[b];
    #pragma unroll
    for (int s = 0; s < 4; ++s) {
      int col = tileN + wc * 64 + s * 16 + r;
      float lx  = act[b * 2048 + 1536 + col];
      float blc = b_lc[col];
      float denomE = 0.0f, numer = 0.0f;
      #pragma unroll
      for (int j = 0; j < 8; ++j) {            // acc row j == word slot j for batch b
        if (j < nw) {
          float e = __expf(acc[i][s][j] + lx + blc);
          denomE += e;
          numer  += e * cbew[((b << 3) + j) * 512 + col];
        }
      }
      float iv = act[b * 2048 + col];
      float g  = act[b * 2048 + 512 + col];
      float o  = act[b * 2048 + 1024 + col];
      float ei = __expf(iv);
      float cv = (nw > 0) ? (numer + ei * g) / (ei + denomE) : g;
      float h1 = o * tanh_fast(cv);
      out[b * 512 + col] = h1;
      out[(long)8192 * 512 + b * 512 + col] = cv;
    }
  }
}

// ---------- launcher ----------
extern "C" void kernel_launch(void* const* d_in, const int* in_sizes, int n_in,
                              void* d_out, int out_size, void* d_ws, size_t ws_size,
                              hipStream_t stream) {
  (void)in_sizes; (void)n_in; (void)out_size; (void)ws_size;
  const float* x     = (const float*)d_in[0];
  const float* h0    = (const float*)d_in[1];
  const float* cbew  = (const float*)d_in[3];
  const int*   nword = (const int*)d_in[4];
  const float* w_ii = (const float*)d_in[5];  const float* b_ii = (const float*)d_in[6];
  const float* w_hi = (const float*)d_in[7];  const float* b_hi = (const float*)d_in[8];
  const float* w_ig = (const float*)d_in[13]; const float* b_ig = (const float*)d_in[14];
  const float* w_hg = (const float*)d_in[15]; const float* b_hg = (const float*)d_in[16];
  const float* w_io = (const float*)d_in[17]; const float* b_io = (const float*)d_in[18];
  const float* w_ho = (const float*)d_in[19]; const float* b_ho = (const float*)d_in[20];
  const float* w_lx = (const float*)d_in[21]; const float* b_lx = (const float*)d_in[22];
  const float* w_lc = (const float*)d_in[23]; const float* b_lc = (const float*)d_in[24];

  char* ws = (char*)d_ws;
  unsigned short* Wxt  = (unsigned short*)(ws + 0);          // 2048*320*2  = 1310720
  unsigned short* Wht  = (unsigned short*)(ws + 1310720);    // 2048*512*2  = 2097152
  unsigned short* Wlct = (unsigned short*)(ws + 3407872);    // 512*512*2   = 524288
  float*          bias = (float*)(ws + 3932160);             // 2048*4      = 8192
  unsigned short* xb   = (unsigned short*)(ws + 3940352);    // 8192*320*2  = 5242880
  unsigned short* hb   = (unsigned short*)(ws + 9183232);    // 8192*512*2  = 8388608
  unsigned short* cbB  = (unsigned short*)(ws + 17571840);   // 65536*512*2 = 67108864
  float*          act  = (float*)(ws + 84680704);            // 8192*2048*4 = 67108864
  float* out = (float*)d_out;

  pack_wx  <<<(2048 * 320 + 255) / 256, 256, 0, stream>>>(w_ii, w_ig, w_io, w_lx, Wxt);
  pack_wh  <<<(2048 * 512 + 255) / 256, 256, 0, stream>>>(w_hi, w_hg, w_ho, Wht);
  pack_wlc <<<(512 * 512 + 255) / 256, 256, 0, stream>>>(w_lc, Wlct);
  pack_bias<<<8, 256, 0, stream>>>(b_ii, b_hi, b_ig, b_hg, b_io, b_ho, b_lx, bias);
  conv_pad_x<<<(8192 * 320 + 255) / 256, 256, 0, stream>>>(x, xb);
  conv_bf16x2<<<(int)((8192L * 512 / 2 + 255) / 256), 256, 0, stream>>>(
      h0, (unsigned int*)hb, 8192L * 512 / 2);
  conv_bf16x2<<<(int)((65536L * 512 / 2 + 255) / 256), 256, 0, stream>>>(
      cbew, (unsigned int*)cbB, 65536L * 512 / 2);

  gate_kernel<<<dim3(16, 64), 256, 0, stream>>>(xb, hb, Wxt, Wht, bias, act);
  edge_kernel<<<dim3(4, 512), 256, 0, stream>>>(cbB, cbew, Wlct, act, b_lc, nword, out);
}
